// VRWKV_Rec_HW_SpatialMix_12249246728670
// MI455X (gfx1250) — matrix operand order repaired and checked by reading, not compiled, not run on hardware
//
#include <hip/hip_runtime.h>
#include <hip/hip_bf16.h>
#include <stdint.h>

#define Bz 16
#define Cc 256
#define Hh 64
#define Ww 64
#define Tt (Hh * Ww)        // 4096
#define Mrows (Bz * Tt)     // 65536
#define SEG 64
#define NSEG (Tt / SEG)     // 64
#define BCP (Bz * Cc)       // 4096
#define AGGSTRIDE (NSEG * BCP)  // 262144

typedef __attribute__((ext_vector_type(16))) __bf16 v16bf;
typedef __attribute__((ext_vector_type(8)))  __bf16 v8bf;
typedef __attribute__((ext_vector_type(8)))  float  v8f;

__device__ __forceinline__ unsigned short f2bf_bits(float f) {
  unsigned u = __float_as_uint(f);
  return (unsigned short)((u + 0x7FFFu + ((u >> 16) & 1u)) >> 16);  // RNE
}
__device__ __forceinline__ __bf16 bfc(float f) {
  return __builtin_bit_cast(__bf16, f2bf_bits(f));
}
__device__ __forceinline__ float fast_rcp(float x) {
  return __builtin_amdgcn_rcpf(x);  // v_rcp_f32, 1 ulp — fine for positive sums
}

// ---------------------------------------------------------------------------
// Prep: fused 5x5 depthwise stencil table + WMMA-fragment-swizzled bf16 weights
//   weff  : [25][256] f32
//   wcat  : [Wk|Wv|Wr] bf16, B-fragment layout [nT][kT][lane][16]
//   wof   : Wo bf16, same fragment layout
// ---------------------------------------------------------------------------
__global__ void __launch_bounds__(256)
prep_kernel(const float* __restrict__ alpha, const float* __restrict__ w1,
            const float* __restrict__ w3, const float* __restrict__ w5,
            const float* __restrict__ Wk, const float* __restrict__ Wv,
            const float* __restrict__ Wr, const float* __restrict__ Wo,
            float* __restrict__ weff, unsigned short* __restrict__ wcat,
            unsigned short* __restrict__ wof) {
  const int idx = blockIdx.x * 256 + threadIdx.x;
  if (idx < 25 * Cc) {                       // fused stencil table
    const int i = idx / Cc, c = idx % Cc;
    const int dy = i / 5, dx = i % 5;
    float v = alpha[3] * w5[c * 25 + i];
    if (dy >= 1 && dy <= 3 && dx >= 1 && dx <= 3)
      v += alpha[2] * w3[c * 9 + (dy - 1) * 3 + (dx - 1)];
    if (i == 12) v += alpha[1] * w1[c] + alpha[0];
    weff[i * Cc + c] = v;
  }
  const int e1 = idx - 25 * Cc;              // Wk|Wv|Wr fragments (N=768)
  if (e1 >= 0 && e1 < 768 * Cc) {
    const int el = e1 & 15, lane = (e1 >> 4) & 31, kT = (e1 >> 9) & 7, nT = e1 >> 12;
    const int K = kT * 32 + ((lane >> 4) << 3) + (el & 7) + ((el >> 3) << 4);
    const int N = nT * 16 + (lane & 15);
    const int seg = N >> 8, col = N & 255;
    const float* Wsrc = (seg == 0) ? Wk : (seg == 1 ? Wv : Wr);
    wcat[e1] = f2bf_bits(Wsrc[K * Cc + col]);
  }
  const int e2 = idx - 25 * Cc - 768 * Cc;   // Wo fragments (N=256)
  if (e2 >= 0 && e2 < 256 * Cc) {
    const int el = e2 & 15, lane = (e2 >> 4) & 31, kT = (e2 >> 9) & 7, nT = e2 >> 12;
    const int K = kT * 32 + ((lane >> 4) << 3) + (el & 7) + ((el >> 3) << 4);
    const int N = nT * 16 + (lane & 15);
    wof[e2] = f2bf_bits(Wo[K * Cc + N]);
  }
}

// ---------------------------------------------------------------------------
// omni_shift: single fused 5x5 depthwise conv (NHWC, lane = channel), bf16 out
// ---------------------------------------------------------------------------
__global__ void __launch_bounds__(256)
omni_shift(const float* __restrict__ X, const float* __restrict__ weff,
           unsigned short* __restrict__ XS) {
  __shared__ float sw[25 * Cc];
  for (int i = threadIdx.x; i < 25 * Cc; i += 256) sw[i] = weff[i];
  __syncthreads();
  const int pix = blockIdx.x;               // [0, B*H*W)
  const int c = threadIdx.x;
  const int b = pix >> 12, hw = pix & (Tt - 1);
  const int h = hw >> 6, w = hw & 63;
  const float* xb = X + (size_t)b * Tt * Cc;
  float acc = 0.0f;
#pragma unroll
  for (int dy = 0; dy < 5; ++dy) {
    const int yy = h + dy - 2;
    if ((unsigned)yy >= (unsigned)Hh) continue;
#pragma unroll
    for (int dx = 0; dx < 5; ++dx) {
      const int xx = w + dx - 2;
      if ((unsigned)xx >= (unsigned)Ww) continue;
      acc += xb[((size_t)(yy * Ww + xx)) * Cc + c] * sw[(dy * 5 + dx) * Cc + c];
    }
  }
  XS[(size_t)pix * Cc + c] = f2bf_bits(acc);
}

// ---------------------------------------------------------------------------
// WMMA bf16 GEMM #1: [65536,256] x [256,768] -> k, v, sigmoid->sr (f32)
// Wave computes 32x32 via 2x2 v_wmma_f32_16x16x32_bf16 tiles, K-loop 8 steps.
// ---------------------------------------------------------------------------
__device__ __forceinline__ void store_kvr(v8f acc, int mbase, int ncol,
                                          float* Ko, float* Vo, float* So) {
  const int seg = ncol >> 8, col = ncol & 255;
  float* dst = (seg == 0) ? Ko : (seg == 1 ? Vo : So);
#pragma unroll
  for (int r = 0; r < 8; ++r) {
    float v = acc[r];
    if (seg == 2) v = fast_rcp(1.0f + __expf(-v));   // sigmoid via v_rcp_f32
    dst[(size_t)(mbase + r) * Cc + col] = v;
  }
}

__global__ void __launch_bounds__(256)
gemm_kvr(const unsigned short* __restrict__ A, const unsigned short* __restrict__ Wf,
         float* __restrict__ Ko, float* __restrict__ Vo, float* __restrict__ So) {
  const int lane = threadIdx.x & 31;
  const int wave = threadIdx.x >> 5;
  const int m0 = blockIdx.x * 256 + wave * 32;
  const int n0 = blockIdx.y * 32;
  const int rowlo = lane & 15;
  const int khi = (lane >> 4) << 3;  // lanes 0-15: K {0..7,16..23}; 16-31: {8..15,24..31}
  v8f acc00 = {}, acc01 = {}, acc10 = {}, acc11 = {};
  for (int kk = 0; kk < Cc; kk += 32) {
    v16bf afrag[2];
#pragma unroll
    for (int i = 0; i < 2; ++i) {
      const unsigned short* ap = A + (size_t)(m0 + i * 16 + rowlo) * Cc + kk + khi;
      __builtin_prefetch(ap + 32, 0, 3);  // speculative global_prefetch_b8, branch-free
      v8bf lo = *(const v8bf*)ap;
      v8bf hi = *(const v8bf*)(ap + 16);
#pragma unroll
      for (int e = 0; e < 8; ++e) { afrag[i][e] = lo[e]; afrag[i][8 + e] = hi[e]; }
    }
    v16bf bfrag[2];
#pragma unroll
    for (int jj = 0; jj < 2; ++jj) {
      const int nT = (n0 >> 4) + jj;
      bfrag[jj] = *(const v16bf*)(Wf + (((size_t)nT * 8 + (kk >> 5)) * 32 + lane) * 16);
    }
    acc00 = __builtin_amdgcn_wmma_f32_16x16x32_bf16(false, afrag[0], false, bfrag[0], (short)0, acc00, false, false);
    acc01 = __builtin_amdgcn_wmma_f32_16x16x32_bf16(false, afrag[0], false, bfrag[1], (short)0, acc01, false, false);
    acc10 = __builtin_amdgcn_wmma_f32_16x16x32_bf16(false, afrag[1], false, bfrag[0], (short)0, acc10, false, false);
    acc11 = __builtin_amdgcn_wmma_f32_16x16x32_bf16(false, afrag[1], false, bfrag[1], (short)0, acc11, false, false);
  }
  const int mlo = m0 + ((lane >> 4) << 3);
  const int nlo = n0 + (lane & 15);
  store_kvr(acc00, mlo,      nlo,      Ko, Vo, So);
  store_kvr(acc01, mlo,      nlo + 16, Ko, Vo, So);
  store_kvr(acc10, mlo + 16, nlo,      Ko, Vo, So);
  store_kvr(acc11, mlo + 16, nlo + 16, Ko, Vo, So);
}

// ---------------------------------------------------------------------------
// Segmented bi-WKV scan.  State triple (a,b,p) with stabilized log-sum-exp
// decay recurrence; segment combine (assoc.): p'=max(p+L*w,P),
// a'=e^{p+Lw-p'}a + e^{P-p'}A  (same for b).
// ---------------------------------------------------------------------------
__device__ __forceinline__ int smap(int s, int colmajor) {
  return colmajor ? ((s & (Hh - 1)) * Ww + (s >> 6)) : s;  // h-major traversal
}

// Phase 1: per-64-element segment aggregates, both directions. 262144 threads.
__global__ void __launch_bounds__(256)
wkv_seg_agg(const float* __restrict__ Kb, const float* __restrict__ Vb,
            const float* __restrict__ sdecay, int j, int colmajor,
            float* __restrict__ aggF, float* __restrict__ aggB) {
  const int gid = blockIdx.x * 256 + threadIdx.x;
  const int c = gid & 255, b = (gid >> 8) & 15, seg = gid >> 12;
  const float wneg = -__expf(sdecay[j * Cc + c] * (1.0f / (float)Tt));
  const size_t base = (size_t)b * Tt * Cc + c;
  const size_t idx = (size_t)seg * BCP + (b << 8) + c;
  float a = 0.f, bb = 0.f, p = -1e38f;
  for (int i = 0; i < SEG; ++i) {
    const size_t off = base + (size_t)smap(seg * SEG + i, colmajor) * Cc;
    const float kt = Kb[off], vt = Vb[off];
    const float q = fmaxf(p + wneg, kt);
    const float e1 = __expf(p + wneg - q), e2 = __expf(kt - q);
    a = e1 * a + e2 * vt; bb = e1 * bb + e2; p = q;
  }
  aggF[idx] = a; aggF[idx + AGGSTRIDE] = bb; aggF[idx + 2 * AGGSTRIDE] = p;
  a = 0.f; bb = 0.f; p = -1e38f;
  for (int i = SEG - 1; i >= 0; --i) {
    const size_t off = base + (size_t)smap(seg * SEG + i, colmajor) * Cc;
    const float kt = Kb[off], vt = Vb[off];
    const float q = fmaxf(p + wneg, kt);
    const float e1 = __expf(p + wneg - q), e2 = __expf(kt - q);
    a = e1 * a + e2 * vt; bb = e1 * bb + e2; p = q;
  }
  aggB[idx] = a; aggB[idx + AGGSTRIDE] = bb; aggB[idx + 2 * AGGSTRIDE] = p;
}

// Phase 2: per-(b,c) exclusive scan over 64 segment aggregates -> seeds in place.
__global__ void __launch_bounds__(256)
wkv_seg_scan(const float* __restrict__ sdecay, int j,
             float* __restrict__ aggF, float* __restrict__ aggB) {
  const int g = blockIdx.x * 256 + threadIdx.x;  // [0, B*C)
  const int c = g & 255;
  const float wneg = -__expf(sdecay[j * Cc + c] * (1.0f / (float)Tt));
  const float dec = wneg * (float)SEG;
  float ca = 0.f, cb = 0.f, cp = -1e38f;
  for (int s = 0; s < NSEG; ++s) {
    const size_t idx = (size_t)s * BCP + g;
    const float A = aggF[idx], Bv = aggF[idx + AGGSTRIDE], P = aggF[idx + 2 * AGGSTRIDE];
    aggF[idx] = ca; aggF[idx + AGGSTRIDE] = cb; aggF[idx + 2 * AGGSTRIDE] = cp;
    const float np = fmaxf(cp + dec, P);
    const float e1 = __expf(cp + dec - np), e2 = __expf(P - np);
    ca = e1 * ca + e2 * A; cb = e1 * cb + e2 * Bv; cp = np;
  }
  ca = 0.f; cb = 0.f; cp = -1e38f;
  for (int s = NSEG - 1; s >= 0; --s) {
    const size_t idx = (size_t)s * BCP + g;
    const float A = aggB[idx], Bv = aggB[idx + AGGSTRIDE], P = aggB[idx + 2 * AGGSTRIDE];
    aggB[idx] = ca; aggB[idx + AGGSTRIDE] = cb; aggB[idx + 2 * AGGSTRIDE] = cp;
    const float np = fmaxf(cp + dec, P);
    const float e1 = __expf(cp + dec - np), e2 = __expf(P - np);
    ca = e1 * ca + e2 * A; cb = e1 * cb + e2 * Bv; cp = np;
  }
}

// Phase 3: replay each segment, staging per-t backward states in 192KB LDS,
// then forward walk combining fwd/bwd/self terms; v updated in place.
__global__ void __launch_bounds__(256)
wkv_combine(const float* __restrict__ Kb, float* __restrict__ Vb,
            const float* __restrict__ sdecay, const float* __restrict__ sfirst,
            int j, int colmajor,
            const float* __restrict__ aggF, const float* __restrict__ aggB) {
  extern __shared__ float smem[];
  float* sA = smem;
  float* sB = smem + SEG * 256;
  float* sP = smem + 2 * SEG * 256;
  const int tid = threadIdx.x;
  const int gid = blockIdx.x * 256 + tid;
  const int c = gid & 255, b = (gid >> 8) & 15, seg = gid >> 12;
  const float invT = 1.0f / (float)Tt;
  const float wneg = -__expf(sdecay[j * Cc + c] * invT);
  const float u = sfirst[j * Cc + c] * invT;
  const size_t base = (size_t)b * Tt * Cc + c;
  const size_t idx = (size_t)seg * BCP + (b << 8) + c;
  float a = aggB[idx], bb = aggB[idx + AGGSTRIDE], p = aggB[idx + 2 * AGGSTRIDE];
  for (int i = SEG - 1; i >= 0; --i) {
    const size_t off = base + (size_t)smap(seg * SEG + i, colmajor) * Cc;
    const float kt = Kb[off], vt = Vb[off];
    sA[i * 256 + tid] = a; sB[i * 256 + tid] = bb; sP[i * 256 + tid] = p;
    const float q = fmaxf(p + wneg, kt);
    const float e1 = __expf(p + wneg - q), e2 = __expf(kt - q);
    a = e1 * a + e2 * vt; bb = e1 * bb + e2; p = q;
  }
  float af = aggF[idx], bf = aggF[idx + AGGSTRIDE], pf = aggF[idx + 2 * AGGSTRIDE];
  for (int i = 0; i < SEG; ++i) {
    const size_t off = base + (size_t)smap(seg * SEG + i, colmajor) * Cc;
    const float kt = Kb[off], vt = Vb[off];
    const float ab = sA[i * 256 + tid], b2 = sB[i * 256 + tid], pb = sP[i * 256 + tid];
    const float ps = u + kt;
    const float q = fmaxf(fmaxf(pf, pb), ps);
    const float ef = __expf(pf - q), eb = __expf(pb - q), es = __expf(ps - q);
    const float num = ef * af + eb * ab + es * vt;
    const float den = ef * bf + eb * b2 + es;
    Vb[off] = num * fast_rcp(den);              // den > 0; v_rcp_f32 + mul
    const float q2 = fmaxf(pf + wneg, kt);
    const float e1 = __expf(pf + wneg - q2), e2 = __expf(kt - q2);
    af = e1 * af + e2 * vt; bf = e1 * bf + e2; pf = q2;
  }
}

// ---------------------------------------------------------------------------
// WMMA bf16 GEMM #2: out = (sr * v) @ Wo, A built on the fly (f32 -> bf16)
// ---------------------------------------------------------------------------
__global__ void __launch_bounds__(256)
gemm_out(const float* __restrict__ Sr, const float* __restrict__ Vv,
         const unsigned short* __restrict__ Wf, float* __restrict__ Out) {
  const int lane = threadIdx.x & 31;
  const int wave = threadIdx.x >> 5;
  const int m0 = blockIdx.x * 256 + wave * 32;
  const int n0 = blockIdx.y * 32;
  const int rowlo = lane & 15;
  const int khi = (lane >> 4) << 3;
  v8f acc00 = {}, acc01 = {}, acc10 = {}, acc11 = {};
  for (int kk = 0; kk < Cc; kk += 32) {
    v16bf afrag[2];
#pragma unroll
    for (int i = 0; i < 2; ++i) {
      const size_t rb = (size_t)(m0 + i * 16 + rowlo) * Cc + kk + khi;
#pragma unroll
      for (int e = 0; e < 8; ++e) {
        afrag[i][e]     = bfc(Sr[rb + e]      * Vv[rb + e]);
        afrag[i][8 + e] = bfc(Sr[rb + 16 + e] * Vv[rb + 16 + e]);
      }
    }
    v16bf bfrag[2];
#pragma unroll
    for (int jj = 0; jj < 2; ++jj) {
      const int nT = (n0 >> 4) + jj;
      bfrag[jj] = *(const v16bf*)(Wf + (((size_t)nT * 8 + (kk >> 5)) * 32 + lane) * 16);
    }
    acc00 = __builtin_amdgcn_wmma_f32_16x16x32_bf16(false, afrag[0], false, bfrag[0], (short)0, acc00, false, false);
    acc01 = __builtin_amdgcn_wmma_f32_16x16x32_bf16(false, afrag[0], false, bfrag[1], (short)0, acc01, false, false);
    acc10 = __builtin_amdgcn_wmma_f32_16x16x32_bf16(false, afrag[1], false, bfrag[0], (short)0, acc10, false, false);
    acc11 = __builtin_amdgcn_wmma_f32_16x16x32_bf16(false, afrag[1], false, bfrag[1], (short)0, acc11, false, false);
  }
  const int mlo = m0 + ((lane >> 4) << 3);
  const int nlo = n0 + (lane & 15);
#pragma unroll
  for (int r = 0; r < 8; ++r) {
    Out[(size_t)(mlo + r) * Cc + nlo]           = acc00[r];
    Out[(size_t)(mlo + r) * Cc + nlo + 16]      = acc01[r];
    Out[(size_t)(mlo + 16 + r) * Cc + nlo]      = acc10[r];
    Out[(size_t)(mlo + 16 + r) * Cc + nlo + 16] = acc11[r];
  }
}

// ---------------------------------------------------------------------------
extern "C" void kernel_launch(void* const* d_in, const int* in_sizes, int n_in,
                              void* d_out, int out_size, void* d_ws, size_t ws_size,
                              hipStream_t stream) {
  (void)in_sizes; (void)n_in; (void)out_size; (void)ws_size;
  const float* x      = (const float*)d_in[0];
  const float* alpha  = (const float*)d_in[3];
  const float* w1     = (const float*)d_in[4];
  const float* w3     = (const float*)d_in[5];
  const float* w5     = (const float*)d_in[6];
  const float* Wk     = (const float*)d_in[7];
  const float* Wv     = (const float*)d_in[8];
  const float* Wr     = (const float*)d_in[9];
  const float* Wo     = (const float*)d_in[10];
  const float* sdecay = (const float*)d_in[11];
  const float* sfirst = (const float*)d_in[12];
  char* ws = (char*)d_ws;

  float*          weff  = (float*)(ws + 0);                  //  25.6 KB
  unsigned short* wcat  = (unsigned short*)(ws + 25600);     // 384 KB
  unsigned short* wof   = (unsigned short*)(ws + 418816);    // 128 KB
  unsigned short* xs    = (unsigned short*)(ws + 549888);    //  32 MB bf16
  float*          kbuf  = (float*)(ws + 34104320);           //  64 MB
  float*          vbuf  = (float*)(ws + 101213184);          //  64 MB
  float*          srbuf = (float*)(ws + 168322048);          //  64 MB
  float*          aggF  = (float*)(ws + 235430912);          //   3 MB
  float*          aggB  = (float*)(ws + 238576640);          //   3 MB
  float* out = (float*)d_out;

  prep_kernel<<<1049, 256, 0, stream>>>(alpha, w1, w3, w5, Wk, Wv, Wr, Wo,
                                        weff, wcat, wof);
  omni_shift<<<Bz * Tt, 256, 0, stream>>>(x, weff, xs);
  gemm_kvr<<<dim3(Mrows / 256, 24), 256, 0, stream>>>(xs, wcat, kbuf, vbuf, srbuf);
  for (int j = 0; j < 2; ++j) {
    const int cm = j & 1;  // j odd -> column-major (W-scan) traversal
    wkv_seg_agg<<<1024, 256, 0, stream>>>(kbuf, vbuf, sdecay, j, cm, aggF, aggB);
    wkv_seg_scan<<<16, 256, 0, stream>>>(sdecay, j, aggF, aggB);
    wkv_combine<<<1024, 256, SEG * 256 * 3 * sizeof(float), stream>>>(
        kbuf, vbuf, sdecay, sfirst, j, cm, aggF, aggB);
  }
  gemm_out<<<dim3(Mrows / 256, 8), 256, 0, stream>>>(srbuf, vbuf, wof, out);
}
